// ContrastiveDist_42013370089993
// MI455X (gfx1250) — matrix operand order repaired and checked by reading, not compiled, not run on hardware
//
#include <hip/hip_runtime.h>
#include <hip/hip_bf16.h>

// CDNA5 / gfx1250: wave32, WMMA 16x16x32 bf16 path.
typedef __attribute__((ext_vector_type(16))) __bf16 v16bf;
typedef __attribute__((ext_vector_type(8)))  __bf16 v8bf;
typedef __attribute__((ext_vector_type(8)))  float  v8f;

#define CD_EPS 1e-8f

// ---------------------------------------------------------------------------
// Prep: split target (E x 256, f32) into row-major bf16 hi/lo halves and
// compute tn[e] = ||target[e]||. One 256-thread block per row.
// ---------------------------------------------------------------------------
__global__ __launch_bounds__(256)
void ContrastiveDist_prep_kernel(const float* __restrict__ target,
                                 __bf16* __restrict__ ahi,
                                 __bf16* __restrict__ alo,
                                 float* __restrict__ tn) {
    const int row = blockIdx.x;
    const int tid = threadIdx.x;               // D == 256
    const int idx = row * 256 + tid;
    float x = target[idx];
    __bf16 h = (__bf16)x;
    ahi[idx] = h;
    alo[idx] = (__bf16)(x - (float)h);

    __shared__ float red[256];
    red[tid] = x * x;
    __syncthreads();
    for (int s = 128; s > 0; s >>= 1) {
        if (tid < s) red[tid] += red[tid + s];
        __syncthreads();
    }
    if (tid == 0) tn[row] = sqrtf(red[0]);
}

// ---------------------------------------------------------------------------
// Main: one wave per 16-column n-tile. B fragments (node_emb hi/lo bf16)
// register-resident; loop over all e-tiles with 3x v_wmma_f32_16x16x32_bf16
// per K=32 block (hi*hi + hi*lo + lo*hi), fused scale + column reduction.
// ---------------------------------------------------------------------------
__global__ __launch_bounds__(128)
void ContrastiveDist_main_kernel(const float* __restrict__ node_emb,
                                 const __bf16* __restrict__ ahi,
                                 const __bf16* __restrict__ alo,
                                 const float* __restrict__ tn,
                                 float* __restrict__ out,
                                 int E, int N) {
    const int lane = threadIdx.x & 31;
    const int wave = threadIdx.x >> 5;
    const int tile = blockIdx.x * 4 + wave;
    const int ntiles = N >> 4;
    if (tile >= ntiles) return;                // wave-uniform exit; EXEC all-1s for WMMA

    const int n0   = tile << 4;
    const int ncol = lane & 15;
    const int half = lane >> 4;                // 0: lanes 0-15, 1: lanes 16-31

    // ---- Load B tile (16 node_emb rows, K=0..255), convert to bf16 hi/lo.
    // B k order used here (contiguous, not the ISA table order): lanes 0-15 hold K=k0..k0+15 of column n, lanes 16-31
    // hold K=k0+16..k0+31  -> each lane reads a contiguous run of 16 f32.
    v16bf bhi[8], blo[8];
    float sumsq = 0.f;
    const float* brow = node_emb + (size_t)(n0 + ncol) * 256 + half * 16;
#pragma unroll
    for (int kb = 0; kb < 8; ++kb) {
        union { v16bf v; __bf16 e[16]; } uh, ul;
        const float4* bp = (const float4*)(brow + kb * 32);
#pragma unroll
        for (int q = 0; q < 4; ++q) {
            float4 f = bp[q];
            float xs[4] = {f.x, f.y, f.z, f.w};
#pragma unroll
            for (int t = 0; t < 4; ++t) {
                float x = xs[t];
                __bf16 h = (__bf16)x;
                uh.e[q * 4 + t] = h;
                ul.e[q * 4 + t] = (__bf16)(x - (float)h);
                sumsq += x * x;                // exact f32 for ||v_n||
            }
        }
        bhi[kb] = uh.v;
        blo[kb] = ul.v;
    }
    // lanes n and n+16 each hold half the K values -> combine across halves
    sumsq += __shfl_xor(sumsq, 16, 32);
    const float vn = sqrtf(sumsq);

    float colsum = 0.f;
    const int etiles = E >> 4;
    for (int et = 0; et < etiles; ++et) {
        // A 16x32 bf16 layout: lane m (per half) holds K run [k0+16*half, +16)
        // (matches B) -> two contiguous 16B loads per matrix.
        const __bf16* arh = ahi + (size_t)(et * 16 + ncol) * 256 + half * 16;
        const __bf16* arl = alo + (size_t)(et * 16 + ncol) * 256 + half * 16;

        v8f acc = {0.f, 0.f, 0.f, 0.f, 0.f, 0.f, 0.f, 0.f};
#pragma unroll
        for (int kb = 0; kb < 8; ++kb) {
            v8bf h0 = *(const v8bf*)(arh + kb * 32);
            v8bf h1 = *(const v8bf*)(arh + kb * 32 + 8);
            v8bf l0 = *(const v8bf*)(arl + kb * 32);
            v8bf l1 = *(const v8bf*)(arl + kb * 32 + 8);
            v16bf a_hi = __builtin_shufflevector(h0, h1, 0,1,2,3,4,5,6,7,8,9,10,11,12,13,14,15);
            v16bf a_lo = __builtin_shufflevector(l0, l1, 0,1,2,3,4,5,6,7,8,9,10,11,12,13,14,15);
            acc = __builtin_amdgcn_wmma_f32_16x16x32_bf16(false, a_hi, false, bhi[kb],
                                                          (short)0, acc, false, false);
            acc = __builtin_amdgcn_wmma_f32_16x16x32_bf16(false, a_hi, false, blo[kb],
                                                          (short)0, acc, false, false);
            acc = __builtin_amdgcn_wmma_f32_16x16x32_bf16(false, a_lo, false, bhi[kb],
                                                          (short)0, acc, false, false);
        }
        // C layout: lane holds column n0+ncol, rows m = 8*half + j (j=0..7)
        const float4* tp = (const float4*)(tn + et * 16 + half * 8);
        float4 t0 = tp[0];
        float4 t1 = tp[1];
        colsum += acc[0] * __builtin_amdgcn_rcpf(t0.x * vn + CD_EPS);
        colsum += acc[1] * __builtin_amdgcn_rcpf(t0.y * vn + CD_EPS);
        colsum += acc[2] * __builtin_amdgcn_rcpf(t0.z * vn + CD_EPS);
        colsum += acc[3] * __builtin_amdgcn_rcpf(t0.w * vn + CD_EPS);
        colsum += acc[4] * __builtin_amdgcn_rcpf(t1.x * vn + CD_EPS);
        colsum += acc[5] * __builtin_amdgcn_rcpf(t1.y * vn + CD_EPS);
        colsum += acc[6] * __builtin_amdgcn_rcpf(t1.z * vn + CD_EPS);
        colsum += acc[7] * __builtin_amdgcn_rcpf(t1.w * vn + CD_EPS);
    }

    float total = colsum + __shfl_xor(colsum, 16, 32);
    if (half == 0) out[n0 + ncol] = -total;
}

// ---------------------------------------------------------------------------
extern "C" void kernel_launch(void* const* d_in, const int* in_sizes, int n_in,
                              void* d_out, int out_size, void* d_ws, size_t ws_size,
                              hipStream_t stream) {
    (void)n_in; (void)out_size; (void)ws_size;
    // inputs: 0=pred (unused), 1=target [E,256] f32, 2=node_emb [N,256] f32
    const float* target   = (const float*)d_in[1];
    const float* node_emb = (const float*)d_in[2];
    float* out = (float*)d_out;

    const int D = 256;
    const int E = in_sizes[1] / D;   // 2048
    const int N = in_sizes[2] / D;   // 50000

    // workspace: [ahi: E*256 bf16][alo: E*256 bf16][tn: E f32]  (~2.1 MB)
    __bf16* ahi = (__bf16*)d_ws;
    __bf16* alo = (__bf16*)((char*)d_ws + (size_t)E * D * sizeof(__bf16));
    float*  tnp = (float*)((char*)d_ws + (size_t)E * D * 2 * sizeof(__bf16));

    ContrastiveDist_prep_kernel<<<E, 256, 0, stream>>>(target, ahi, alo, tnp);

    const int ntiles = N / 16;                 // 3125
    const int blocks = (ntiles + 3) / 4;       // 4 waves (one tile each) per block
    ContrastiveDist_main_kernel<<<blocks, 128, 0, stream>>>(node_emb, ahi, alo, tnp,
                                                            out, E, N);
}
